// CustomLSTM_10479720202333
// MI455X (gfx1250) — compile-verified, matrix operand order checked by reading, not run on hardware
//
#include <hip/hip_runtime.h>

// ---------------------------------------------------------------------------
// Persistent WMMA LSTM for MI455X (gfx1250, wave32).
//   B=64, T=512, D=512, H=1024, K = D+H = 1536.
//   - 64 workgroups x 128 threads (4 waves). WG w owns H-columns [16w,16w+16)
//     for all 4 gates; wave m owns batch rows [16m,16m+16).
//   - All 4 gate weight slices converted f32->f16 once into LDS (~192 KB/WGP,
//     CDNA5 has 320 KB) -> zero weight traffic for 512 timesteps.
//   - LDS layout is fragment-major [n][kb][gate][32 halfs] (padded n-stride):
//     all 8 B-fragment ds_load_b128 per k-block use immediate offsets off one
//     pointer (+256 B/kb), and b128 reads are bank-conflict free.
//   - x pre-converted to f16 in workspace (if it fits) so every A-fragment is
//     two 16B loads; no per-step cvt VALU on the WMMA critical path.
//   - Cell state c kept in 8 VGPRs (C/D accumulator layout) across all steps.
//   - h double-buffered as f16 in workspace; per-step grid barrier via
//     device-scope atomic counter (release add / acquire spin).
//   - Matrix core: v_wmma_f32_16x16x32_f16, f32 accumulate.
// ---------------------------------------------------------------------------

typedef __attribute__((ext_vector_type(16))) _Float16 v16h;
typedef __attribute__((ext_vector_type(8)))  float    v8f;

#define B_    64
#define T_    512
#define D_    512
#define H_    1024
#define K_    1536
#define NWG   64
#define COLS  16
#define NKB   48                          // k-blocks of 32
// [n][kb][gate][32 halfs]; +8 halfs pad so n-stride = 3076 dwords = 4 mod 64
#define NSTRIDE   (NKB * 4 * 32 + 8)      // 6152 halfs per n
#define LDS_HALFS (COLS * NSTRIDE)        // 98432 halfs
#define LDS_BYTES (LDS_HALFS * 2)         // 196864 B (< 320 KB WGP LDS)

union Frag16 { v16h v; uint4 u[2]; };
union H8     { _Float16 h[8]; uint4 u; };

__device__ __forceinline__ float sigmoidf_fast(float x) {
  return 1.0f / (1.0f + __expf(-x));
}
__device__ __forceinline__ float tanhf_fast(float x) {
  return 2.0f / (1.0f + __expf(-2.0f * x)) - 1.0f;
}

extern __shared__ _Float16 ldsW[];

// Load the 4 B-fragments (one per gate) at p_ (all immediate DS offsets) and
// issue 4 WMMAs. Region layout per k-block: gate g at halfs [g*32, g*32+32).
#define GATE_MMA(p_)                                                             \
  {                                                                              \
    Frag16 b0_, b1_, b2_, b3_;                                                   \
    b0_.u[0] = *(const uint4*)(p_);       b0_.u[1] = *(const uint4*)(p_ + 16);   \
    b1_.u[0] = *(const uint4*)(p_ + 32);  b1_.u[1] = *(const uint4*)(p_ + 48);   \
    b2_.u[0] = *(const uint4*)(p_ + 64);  b2_.u[1] = *(const uint4*)(p_ + 80);   \
    b3_.u[0] = *(const uint4*)(p_ + 96);  b3_.u[1] = *(const uint4*)(p_ + 112);  \
    acc0 = __builtin_amdgcn_wmma_f32_16x16x32_f16(false, a.v, false, b0_.v,      \
                                                  (short)0, acc0, false, false); \
    acc1 = __builtin_amdgcn_wmma_f32_16x16x32_f16(false, a.v, false, b1_.v,      \
                                                  (short)0, acc1, false, false); \
    acc2 = __builtin_amdgcn_wmma_f32_16x16x32_f16(false, a.v, false, b2_.v,      \
                                                  (short)0, acc2, false, false); \
    acc3 = __builtin_amdgcn_wmma_f32_16x16x32_f16(false, a.v, false, b3_.v,      \
                                                  (short)0, acc3, false, false); \
  }

// One-time x f32 -> f16 pass (8 elements / thread, 16B stores).
__global__ __launch_bounds__(256)
void cvt_x_f16(const float* __restrict__ x, _Float16* __restrict__ xh) {
  const size_t i = ((size_t)blockIdx.x * 256 + threadIdx.x) * 8;
  float4 f0 = *(const float4*)(x + i);
  float4 f1 = *(const float4*)(x + i + 4);
  H8 o;
  o.h[0] = (_Float16)f0.x; o.h[1] = (_Float16)f0.y;
  o.h[2] = (_Float16)f0.z; o.h[3] = (_Float16)f0.w;
  o.h[4] = (_Float16)f1.x; o.h[5] = (_Float16)f1.y;
  o.h[6] = (_Float16)f1.z; o.h[7] = (_Float16)f1.w;
  *(uint4*)(xh + i) = o.u;
}

// XF16: x supplied pre-converted to f16 (pure 16B loads in the hot loop).
template <bool XF16>
__global__ __launch_bounds__(128, 1)
void lstm_wmma_persistent(
    const void* __restrict__ xin,
    const float* __restrict__ Wf, const float* __restrict__ bfp,
    const float* __restrict__ Wi, const float* __restrict__ bip,
    const float* __restrict__ Wc, const float* __restrict__ bcp,
    const float* __restrict__ Wo, const float* __restrict__ bop,
    float* __restrict__ out,
    _Float16* __restrict__ hbuf0, _Float16* __restrict__ hbuf1,
    unsigned* __restrict__ counter)
{
  const int wg   = blockIdx.x;        // 0..63: which 16-column slice of H
  const int tid  = threadIdx.x;       // 0..127
  const int lane = tid & 31;
  const int wave = tid >> 5;          // M-tile (batch rows 16*wave..+15)
  const int col0 = wg * COLS;

  // ---- one-time: gate weight slices f32 -> f16 into LDS ----
  // element (g, k, n) -> ldsW[n*NSTRIDE + ((k/32)*4 + g)*32 + (k%32)]
  for (int idx = tid; idx < 4 * COLS * K_; idx += 128) {
    const int n    = idx & 15;
    const int rest = idx >> 4;
    const int k    = rest % K_;
    const int g    = rest / K_;
    const float* W = (g == 0) ? Wf : ((g == 1) ? Wi : ((g == 2) ? Wc : Wo));
    ldsW[n * NSTRIDE + (((k >> 5) * 4 + g) << 5) + (k & 31)] =
        (_Float16)W[(size_t)k * H_ + (col0 + n)];
  }

  const int nb       = lane & 15;     // column within tile / A-row within tile
  const int half_sel = lane >> 4;     // lane-half: k offset +8 (A/B), row +8 (C/D)
  const float bias0 = bfp[col0 + nb];
  const float bias1 = bip[col0 + nb];
  const float bias2 = bcp[col0 + nb];
  const float bias3 = bop[col0 + nb];

  __syncthreads();

  const int brow_a  = wave * 16 + nb;            // batch row this lane loads (A frag)
  const int brow_c0 = wave * 16 + half_sel * 8;  // first batch row this lane owns (C/D)
  const _Float16* lB0 = ldsW + nb * NSTRIDE + half_sel * 8;  // per-lane B base

  v8f cstate = {};                               // c_0 = 0, lives in VGPRs forever

  #pragma unroll 1
  for (int t = 0; t < T_; ++t) {
    const _Float16* hin  = (t & 1) ? hbuf1 : hbuf0;
    _Float16*       hout = (t & 1) ? hbuf0 : hbuf1;

    v8f acc0 = {}, acc1 = {}, acc2 = {}, acc3 = {};

    const _Float16* hrow = hin + (size_t)brow_a * H_;

    // ---- k-blocks 0..15: A from x_t ----
    if (XF16) {
      const _Float16* xrow =
          (const _Float16*)xin + ((size_t)brow_a * T_ + t) * D_;
      for (int kb = 0; kb < 16; ++kb) {
        const int abase = kb * 32 + half_sel * 8;
        Frag16 a;
        a.u[0] = *(const uint4*)(xrow + abase);
        a.u[1] = *(const uint4*)(xrow + abase + 16);
        GATE_MMA(lB0 + kb * 128);
      }
    } else {
      const float* xrow = (const float*)xin + ((size_t)brow_a * T_ + t) * D_;
      for (int kb = 0; kb < 16; ++kb) {
        const int abase = kb * 32 + half_sel * 8;
        Frag16 a;
        float4 f0 = *(const float4*)(xrow + abase);
        float4 f1 = *(const float4*)(xrow + abase + 4);
        float4 f2 = *(const float4*)(xrow + abase + 16);
        float4 f3 = *(const float4*)(xrow + abase + 20);
        a.v[0]  = (_Float16)f0.x; a.v[1]  = (_Float16)f0.y;
        a.v[2]  = (_Float16)f0.z; a.v[3]  = (_Float16)f0.w;
        a.v[4]  = (_Float16)f1.x; a.v[5]  = (_Float16)f1.y;
        a.v[6]  = (_Float16)f1.z; a.v[7]  = (_Float16)f1.w;
        a.v[8]  = (_Float16)f2.x; a.v[9]  = (_Float16)f2.y;
        a.v[10] = (_Float16)f2.z; a.v[11] = (_Float16)f2.w;
        a.v[12] = (_Float16)f3.x; a.v[13] = (_Float16)f3.y;
        a.v[14] = (_Float16)f3.z; a.v[15] = (_Float16)f3.w;
        GATE_MMA(lB0 + kb * 128);
      }
    }

    // ---- k-blocks 16..47: A from h_{t-1} (f16, two 16B loads) ----
    for (int kb = 16; kb < 48; ++kb) {
      const int abase = kb * 32 + half_sel * 8 - D_;
      Frag16 a;
      a.u[0] = *(const uint4*)(hrow + abase);
      a.u[1] = *(const uint4*)(hrow + abase + 16);
      GATE_MMA(lB0 + kb * 128);
    }

    // ---- epilogue: bias + gate nonlinearities + cell update (all in VGPRs) ----
    acc0 += bias0; acc1 += bias1; acc2 += bias2; acc3 += bias3;

    float hv[8];
    #pragma unroll
    for (int r = 0; r < 8; ++r) {
      const float fg = sigmoidf_fast(acc0[r]);
      const float ig = sigmoidf_fast(acc1[r]);
      const float cg = tanhf_fast(acc2[r]);
      const float og = sigmoidf_fast(acc3[r]);
      const float cn = fg * cstate[r] + ig * cg;
      cstate[r] = cn;
      hv[r] = og * tanhf_fast(cn);
    }

    const int colg = col0 + nb;
    if (t == T_ - 1) {
      // final h -> d_out (f32, [B][H])
      #pragma unroll
      for (int r = 0; r < 8; ++r)
        out[(size_t)(brow_c0 + r) * H_ + colg] = hv[r];
    } else {
      // h_t -> f16 double buffer for next step's A-fragments
      #pragma unroll
      for (int r = 0; r < 8; ++r)
        hout[(size_t)(brow_c0 + r) * H_ + colg] = (_Float16)hv[r];

      // ---- grid-wide step barrier (device-scope release/acquire) ----
      __threadfence();          // flush h stores to device scope
      __syncthreads();
      if (tid == 0) {
        __hip_atomic_fetch_add(counter, 1u, __ATOMIC_RELEASE,
                               __HIP_MEMORY_SCOPE_AGENT);
        const unsigned target = (unsigned)(t + 1) * NWG;
        while (__hip_atomic_load(counter, __ATOMIC_ACQUIRE,
                                 __HIP_MEMORY_SCOPE_AGENT) < target) {
          __builtin_amdgcn_s_sleep(1);
        }
      }
      __syncthreads();
    }
  }
}

extern "C" void kernel_launch(void* const* d_in, const int* in_sizes, int n_in,
                              void* d_out, int out_size, void* d_ws, size_t ws_size,
                              hipStream_t stream) {
  (void)in_sizes; (void)n_in; (void)out_size;

  const float* x  = (const float*)d_in[0];
  const float* Wf = (const float*)d_in[1];
  const float* bf = (const float*)d_in[2];
  const float* Wi = (const float*)d_in[3];
  const float* bi = (const float*)d_in[4];
  const float* Wc = (const float*)d_in[5];
  const float* bc = (const float*)d_in[6];
  const float* Wo = (const float*)d_in[7];
  const float* bo = (const float*)d_in[8];
  float* out = (float*)d_out;

  // workspace: [0,128) barrier counter; h double buffers (f16 [B][H] each);
  // then (optional) f16 copy of x.
  const size_t hbytes   = (size_t)B_ * H_ * sizeof(_Float16);      // 128 KB
  const size_t xh_off   = 128 + 2 * hbytes;
  const size_t xh_bytes = (size_t)B_ * T_ * D_ * sizeof(_Float16); // 32 MB
  unsigned* counter = (unsigned*)d_ws;
  _Float16* h0 = (_Float16*)((char*)d_ws + 128);
  _Float16* h1 = h0 + (size_t)B_ * H_;
  _Float16* xh = (_Float16*)((char*)d_ws + xh_off);

  // zero counter + h0 (h1 fully written at t=0 before first read);
  // on-stream so graph replay resets state deterministically.
  hipMemsetAsync(d_ws, 0, 128 + hbytes, stream);

  const bool use_xf16 = ws_size >= xh_off + xh_bytes;

  if (use_xf16) {
    const int nelem  = B_ * T_ * D_;            // 16,777,216
    const int blocks = nelem / (256 * 8);       // 8192
    cvt_x_f16<<<blocks, 256, 0, stream>>>(x, xh);

    hipFuncSetAttribute((const void*)lstm_wmma_persistent<true>,
                        hipFuncAttributeMaxDynamicSharedMemorySize, LDS_BYTES);
    lstm_wmma_persistent<true><<<NWG, 128, LDS_BYTES, stream>>>(
        (const void*)xh, Wf, bf, Wi, bi, Wc, bc, Wo, bo, out, h0, h1, counter);
  } else {
    hipFuncSetAttribute((const void*)lstm_wmma_persistent<false>,
                        hipFuncAttributeMaxDynamicSharedMemorySize, LDS_BYTES);
    lstm_wmma_persistent<false><<<NWG, 128, LDS_BYTES, stream>>>(
        (const void*)x, Wf, bf, Wi, bi, Wc, bc, Wo, bo, out, h0, h1, counter);
  }
}
